// INR_48644799594950
// MI455X (gfx1250) — matrix operand order repaired and checked by reading, not compiled, not run on hardware
//
#include <hip/hip_runtime.h>
#include <hip/hip_bf16.h>

typedef __attribute__((ext_vector_type(16))) _Float16 v16h;
typedef __attribute__((ext_vector_type(8)))  _Float16 v8h;
typedef __attribute__((ext_vector_type(8)))  float    v8f;

// ---- WMMA fragment helpers (CDNA5 16x16x32 f16 layouts, wave32) ----

__device__ __forceinline__ v16h frag_comb(v8h lo, v8h hi) {
    return __builtin_shufflevector(lo, hi, 0,1,2,3,4,5,6,7,8,9,10,11,12,13,14,15);
}

// A matrix 16x32 f16 from row-major LDS stage [16][ldk]:
// lanes 0-15: M=lane, halves 0..7 = K=kb..kb+7, halves 8..15 = K=kb+16..kb+23 (kb=0)
// lanes 16-31: same M, kb=8 offset.
__device__ __forceinline__ v16h loadA(const _Float16* base, int ldk, int lane, int kBase) {
    int row = lane & 15;
    int kb  = kBase + ((lane >> 4) << 3);
    const _Float16* p = base + row * ldk + kb;
    v8h lo = *(const v8h*)(p);
    v8h hi = *(const v8h*)(p + 16);
    return frag_comb(lo, hi);
}

// B matrix 32x16 f16 from column-major LDS [ncols][ldk]:
// lane L holds column colBase+(L&15), K = kb..kb+7 and kb+16..kb+23, kb = kBase+(L>>4)*8.
__device__ __forceinline__ v16h loadB(const _Float16* base, int ldk, int lane, int colBase, int kBase) {
    int col = colBase + (lane & 15);
    int k0  = kBase + ((lane >> 4) << 3);
    const _Float16* p = base + col * ldk + k0;
    v8h lo = *(const v8h*)(p);
    v8h hi = *(const v8h*)(p + 16);
    return frag_comb(lo, hi);
}

__device__ __forceinline__ v8f wmma16(v16h a, v16h b, v8f c) {
    return __builtin_amdgcn_wmma_f32_16x16x32_f16(false, a, false, b, (short)0, c, false, false);
}

#define LDK136 136   // padded leading dim (halves) for 128-K buffers (bank-conflict free, 16B aligned)
#define LDK40  40    // padded leading dim for 32-K buffers
#define TILES1 8     // 16-pixel tiles per wave, kernel 1
#define TILES2 8     // 16-pixel tiles per wave, kernel 2

// ================= Kernel 1: rotate/shift + MLP(2->128->128->3) + color affine =================
__global__ __launch_bounds__(128) void inr_mlp_kernel(
    const float* __restrict__ x, const int* __restrict__ sample_idx,
    const float* __restrict__ shift_vectors, const float* __restrict__ rotation_angle,
    const float* __restrict__ color_w, const float* __restrict__ color_b,
    const float* __restrict__ w1, const float* __restrict__ b1,
    const float* __restrict__ w2, const float* __restrict__ b2,
    const float* __restrict__ w3, const float* __restrict__ b3,
    float* __restrict__ out256)
{
    __shared__ __align__(16) _Float16 w2c[128 * LDK136];   // w2 column-major f16
    __shared__ __align__(16) _Float16 w3c[16 * LDK136];    // w3 padded to 16 cols, column-major f16
    __shared__ __align__(16) float    w1s[256];
    __shared__ float b1s[128];
    __shared__ float b2s[128];
    __shared__ float b3s[4];
    __shared__ __align__(16) _Float16 hstage[4][16 * LDK136]; // per-wave activation stage

    const int tid = threadIdx.x;

    // ---- cooperative weight staging (amortized over TILES1 tiles per wave) ----
    for (int i = tid; i < 128 * 128; i += 128) {
        int k = i >> 7, n = i & 127;           // w2[k][n]
        w2c[n * LDK136 + k] = (_Float16)w2[i];
    }
    for (int i = tid; i < 16 * 128; i += 128) {
        int n = i >> 7, k = i & 127;
        w3c[n * LDK136 + k] = (n < 3) ? (_Float16)w3[k * 3 + n] : (_Float16)0.f;
    }
    for (int i = tid; i < 256; i += 128) w1s[i] = w1[i];
    b1s[tid] = b1[tid];
    b2s[tid] = b2[tid];
    if (tid < 3) b3s[tid] = b3[tid];
    __syncthreads();

    const int wave = tid >> 5, lane = tid & 31;
    const int waveBase = (blockIdx.x * 4 + wave) * (16 * TILES1); // contiguous 128-px span per wave
    const int b = waveBase >> 16;                                 // H*W = 65536 per sample; span never straddles
    const int sidx = sample_idx[b];
    const float shx = shift_vectors[sidx * 2 + 0];
    const float shy = shift_vectors[sidx * 2 + 1];
    const float ang = rotation_angle[sidx];
    const float cs = __cosf(ang), sn = __sinf(ang);
    const int n  = lane & 15;
    const int mo = (lane >> 4) * 8;
    const bool  aff = (sidx != 0);
    float cwv = 0.f, cbv = 0.f, b3v = 0.f;
    if (n < 3) { cwv = color_w[sidx * 3 + n]; cbv = color_b[sidx * 3 + n]; b3v = b3s[n]; }

    _Float16* hs = hstage[wave];

    for (int it = 0; it < TILES1; ++it) {
        const int tileBase = waveBase + it * 16;

        // ---- layer 1 (K=2, VALU) : 2 lanes per pixel, 64 hidden each ----
        {
            const int p  = lane & 15;
            const int jb = (lane >> 4) * 64;
            const int flat = tileBase + p;
            const float X = x[flat * 2 + 0], Y = x[flat * 2 + 1];
            const float cx = cs * X - sn * Y + shx;
            const float cy = sn * X + cs * Y + shy;
            #pragma unroll
            for (int j0 = 0; j0 < 64; j0 += 8) {
                v8h hv;
                #pragma unroll
                for (int u = 0; u < 8; ++u) {
                    int j = jb + j0 + u;
                    float h = fmaf(cx, w1s[j], fmaf(cy, w1s[128 + j], b1s[j]));
                    hv[u] = (_Float16)fmaxf(h, 0.f);
                }
                *(v8h*)(hs + p * LDK136 + jb + j0) = hv;
            }
        }
        asm volatile("s_wait_dscnt 0" ::: "memory");

        // ---- layer 2 (128x128): 32 WMMAs, one-deep software-pipelined fragment loads ----
        v8f acc[8];
        #pragma unroll
        for (int nt = 0; nt < 8; ++nt) acc[nt] = (v8f){};
        {
            v16h a  = loadA(hs, LDK136, lane, 0);
            v16h bf = loadB(w2c, LDK136, lane, 0, 0);
            #pragma unroll
            for (int s = 0; s < 32; ++s) {
                const int nt = s & 7;
                v16h a_n = a, b_n = bf;
                if (s < 31) {
                    const int s2 = s + 1, kc2 = s2 >> 3, nt2 = s2 & 7;
                    if (nt2 == 0) a_n = loadA(hs, LDK136, lane, kc2 * 32);
                    b_n = loadB(w2c, LDK136, lane, nt2 * 16, kc2 * 32);
                }
                acc[nt] = wmma16(a, bf, acc[nt]);
                a = a_n; bf = b_n;
            }
        }

        // ---- bias + relu, restage h2 (overwrites hs; same-wave DS is in-order) ----
        #pragma unroll
        for (int nt = 0; nt < 8; ++nt) {
            const float bb = b2s[nt * 16 + n];
            #pragma unroll
            for (int v = 0; v < 8; ++v) {
                float h = fmaxf(acc[nt][v] + bb, 0.f);
                hs[(mo + v) * LDK136 + nt * 16 + n] = (_Float16)h;
            }
        }
        asm volatile("s_wait_dscnt 0" ::: "memory");

        // ---- layer 3 (128 -> 3, padded to N=16): 4 WMMAs, pipelined ----
        v8f acc3 = (v8f){};
        {
            v16h a  = loadA(hs, LDK136, lane, 0);
            v16h bf = loadB(w3c, LDK136, lane, 0, 0);
            #pragma unroll
            for (int kc = 0; kc < 4; ++kc) {
                v16h a_n = a, b_n = bf;
                if (kc < 3) {
                    a_n = loadA(hs, LDK136, lane, (kc + 1) * 32);
                    b_n = loadB(w3c, LDK136, lane, 0, (kc + 1) * 32);
                }
                acc3 = wmma16(a, bf, acc3);
                a = a_n; bf = b_n;
            }
        }

        if (n < 3) {
            #pragma unroll
            for (int v = 0; v < 8; ++v) {
                float val = acc3[v] + b3v;
                if (aff) val = val * cwv + cbv;
                out256[(tileBase + mo + v) * 3 + n] = val;
            }
        }
        asm volatile("s_wait_dscnt 0" ::: "memory"); // drain before next tile reuses hs
    }
}

// ================= Kernel 2: 2x bilinear upsample + variance head (3->128->3, exp) =================
__global__ __launch_bounds__(128) void inr_up_var_kernel(
    const float* __restrict__ out256,
    const float* __restrict__ vw1, const float* __restrict__ vb1,
    const float* __restrict__ vw2, const float* __restrict__ vb2,
    float* __restrict__ outF, float* __restrict__ varF)
{
    __shared__ __align__(16) _Float16 vw1c[128 * LDK40];   // vw1 padded K=32, column-major
    __shared__ __align__(16) _Float16 vw2c[16 * LDK136];   // vw2 padded to 16 cols, column-major
    __shared__ float vb1s[128];
    __shared__ float vb2s[4];
    __shared__ __align__(16) _Float16 astage[4][16 * LDK40];
    __shared__ __align__(16) _Float16 vstage[4][16 * LDK136];

    const int tid = threadIdx.x;
    for (int i = tid; i < 128 * 32; i += 128) {
        int nn = i >> 5, k = i & 31;
        vw1c[nn * LDK40 + k] = (k < 3) ? (_Float16)vw1[k * 128 + nn] : (_Float16)0.f;
    }
    for (int i = tid; i < 16 * 128; i += 128) {
        int nn = i >> 7, k = i & 127;
        vw2c[nn * LDK136 + k] = (nn < 3) ? (_Float16)vw2[k * 3 + nn] : (_Float16)0.f;
    }
    vb1s[tid] = vb1[tid];
    if (tid < 3) vb2s[tid] = vb2[tid];
    __syncthreads();

    const int wave = tid >> 5, lane = tid & 31;
    const long waveBase = ((long)blockIdx.x * 4 + wave) * (16 * TILES2);
    _Float16* as = astage[wave];
    _Float16* vs = vstage[wave];
    const int p  = lane & 15;
    const int n  = lane & 15;
    const int mo = (lane >> 4) * 8;

    // zero A stage once (K padding; only k<3 slots are rewritten per tile)
    {
        v8h z = (v8h){};
        int p0 = lane >> 1, half = lane & 1;
        *(v8h*)(as + p0 * LDK40 + half * 16 + 0) = z;
        *(v8h*)(as + p0 * LDK40 + half * 16 + 8) = z;
    }

    const int b = (int)(waveBase >> 18);       // 512*512 per sample; wave span never straddles
    const float* src = out256 + ((long)b << 16) * 3;
    const float vb2v = (n < 3) ? vb2s[n] : 0.f;

    for (int it = 0; it < TILES2; ++it) {
        const long tileBase = waveBase + it * 16;
        const long flat2 = tileBase + p;
        const int rem = (int)(flat2 & 262143);
        const int y2 = rem >> 9, x2 = rem & 511;

        // half-pixel-center bilinear, edge-clamped
        const float ty = 0.5f * y2 - 0.25f;
        const float tx = 0.5f * x2 - 0.25f;
        const float fy0 = floorf(ty), fx0 = floorf(tx);
        const float wy = ty - fy0, wx = tx - fx0;
        const int y0 = min(max((int)fy0, 0), 255);
        const int y1 = min(max((int)fy0 + 1, 0), 255);
        const int x0 = min(max((int)fx0, 0), 255);
        const int x1 = min(max((int)fx0 + 1, 0), 255);
        const long r0 = (long)(y0 * 256) * 3, r1 = (long)(y1 * 256) * 3;
        const float w00 = (1.f - wy) * (1.f - wx), w01 = (1.f - wy) * wx;
        const float w10 = wy * (1.f - wx),         w11 = wy * wx;

        // lanes 0-15: channels 0,1 of pixel p ; lanes 16-31: channel 2
        const int c0 = (lane < 16) ? 0 : 2;
        const int nc = (lane < 16) ? 2 : 1;
        for (int q = 0; q < nc; ++q) {
            int c = c0 + q;
            float v00 = src[r0 + x0 * 3 + c], v01 = src[r0 + x1 * 3 + c];
            float v10 = src[r1 + x0 * 3 + c], v11 = src[r1 + x1 * 3 + c];
            float val = w00 * v00 + w01 * v01 + w10 * v10 + w11 * v11;
            outF[flat2 * 3 + c] = val;
            as[p * LDK40 + c] = (_Float16)val;
        }
        asm volatile("s_wait_dscnt 0" ::: "memory");

        // ---- variance layer 1 (3->128, K padded to 32): 8 WMMAs, pipelined B loads ----
        {
            v16h a  = loadA(as, LDK40, lane, 0);
            v16h bf = loadB(vw1c, LDK40, lane, 0, 0);
            #pragma unroll
            for (int nt = 0; nt < 8; ++nt) {
                v16h b_n = bf;
                if (nt < 7) b_n = loadB(vw1c, LDK40, lane, (nt + 1) * 16, 0);
                v8f accv = (v8f){};
                accv = wmma16(a, bf, accv);
                bf = b_n;
                const float bb = vb1s[nt * 16 + n];
                #pragma unroll
                for (int v = 0; v < 8; ++v)
                    vs[(mo + v) * LDK136 + nt * 16 + n] = (_Float16)fmaxf(accv[v] + bb, 0.f);
            }
        }
        asm volatile("s_wait_dscnt 0" ::: "memory");

        // ---- variance layer 2 (128->3 padded to 16): 4 WMMAs, pipelined, then exp ----
        v8f acc3 = (v8f){};
        {
            v16h a  = loadA(vs, LDK136, lane, 0);
            v16h bf = loadB(vw2c, LDK136, lane, 0, 0);
            #pragma unroll
            for (int kc = 0; kc < 4; ++kc) {
                v16h a_n = a, b_n = bf;
                if (kc < 3) {
                    a_n = loadA(vs, LDK136, lane, (kc + 1) * 32);
                    b_n = loadB(vw2c, LDK136, lane, 0, (kc + 1) * 32);
                }
                acc3 = wmma16(a, bf, acc3);
                a = a_n; bf = b_n;
            }
        }
        if (n < 3) {
            #pragma unroll
            for (int v = 0; v < 8; ++v)
                varF[(tileBase + mo + v) * 3 + n] = __expf(acc3[v] + vb2v);
        }
        asm volatile("s_wait_dscnt 0" ::: "memory"); // drain before next tile reuses as/vs
    }
}

// ================= Kernel 3: dx/dy scatter =================
__global__ void inr_tail_kernel(const int* __restrict__ sample_idx,
                                const float* __restrict__ shift_vectors,
                                float* __restrict__ dxp, float* __restrict__ dyp)
{
    int t = threadIdx.x;
    if (t < 8) {
        int s = sample_idx[t];
        dxp[t] = shift_vectors[s * 2 + 0];
        dyp[t] = shift_vectors[s * 2 + 1];
    }
}

extern "C" void kernel_launch(void* const* d_in, const int* in_sizes, int n_in,
                              void* d_out, int out_size, void* d_ws, size_t ws_size,
                              hipStream_t stream) {
    const float* x              = (const float*)d_in[0];
    const int*   sample_idx     = (const int*)d_in[1];
    // d_in[2] = scale_factor (fixed 2), unused
    const float* shift_vectors  = (const float*)d_in[3];
    const float* rotation_angle = (const float*)d_in[4];
    const float* color_w        = (const float*)d_in[5];
    const float* color_b        = (const float*)d_in[6];
    const float* w1  = (const float*)d_in[7];
    const float* b1  = (const float*)d_in[8];
    const float* w2  = (const float*)d_in[9];
    const float* b2  = (const float*)d_in[10];
    const float* w3  = (const float*)d_in[11];
    const float* b3  = (const float*)d_in[12];
    const float* vw1 = (const float*)d_in[13];
    const float* vb1 = (const float*)d_in[14];
    const float* vw2 = (const float*)d_in[15];
    const float* vb2 = (const float*)d_in[16];

    float* out = (float*)d_out;                 // (8,512,512,3)
    float* dx  = out + 6291456;                 // (8,)
    float* dy  = dx + 8;                        // (8,)
    float* var = dy + 8;                        // (8,512,512,3)
    float* out256 = (float*)d_ws;               // scratch: (8,256,256,3) f32 = 6 MB

    // 8*256*256 px / (4 waves * 16 px * 8 tiles) = 1024 blocks
    inr_mlp_kernel<<<1024, 128, 0, stream>>>(x, sample_idx, shift_vectors, rotation_angle,
                                             color_w, color_b, w1, b1, w2, b2, w3, b3, out256);
    // 8*512*512 px / (64 * 8) = 4096 blocks
    inr_up_var_kernel<<<4096, 128, 0, stream>>>(out256, vw1, vb1, vw2, vb2, out, var);
    inr_tail_kernel<<<1, 32, 0, stream>>>(sample_idx, shift_vectors, dx, dy);
}
